// Head_3504693313971
// MI455X (gfx1250) — matrix operand order repaired and checked by reading, compiled, not run on hardware
//
#include <hip/hip_runtime.h>

typedef __attribute__((ext_vector_type(16))) _Float16 v16h;
typedef __attribute__((ext_vector_type(8)))  float    v8f;

#define B_SZ 16
#define T_SZ 2048
#define C_SZ 448
#define H_SZ 64

union F16x8 { uint4 u; _Float16 h[8]; };

#if defined(__has_builtin)
#  if __has_builtin(__builtin_amdgcn_global_load_async_to_lds_b128)
#    define USE_ASYNC_LDS 1
#  endif
#endif

typedef int vi4 __attribute__((vector_size(16)));
typedef __attribute__((address_space(1))) vi4 gas_vi4;   // global
typedef __attribute__((address_space(3))) vi4 las_vi4;   // LDS

// Copy 16 bytes global -> LDS, via gfx1250 async-to-LDS when available.
__device__ __forceinline__ void cp16_g2l(const _Float16* g, _Float16* l) {
#ifdef USE_ASYNC_LDS
    __builtin_amdgcn_global_load_async_to_lds_b128(
        (gas_vi4*)g, (las_vi4*)l, 0, 0);
#else
    *(uint4*)l = *(const uint4*)g;
#endif
}

__device__ __forceinline__ void wait_g2l() {
#ifdef USE_ASYNC_LDS
#  if __has_builtin(__builtin_amdgcn_s_wait_asynccnt)
    __builtin_amdgcn_s_wait_asynccnt(0);
#  else
    asm volatile("s_wait_asynccnt 0" ::: "memory");
#  endif
#endif
}

__device__ __forceinline__ v8f wmma16(v16h a, v16h b, v8f c) {
    return __builtin_amdgcn_wmma_f32_16x16x32_f16(
        false, a, false, b, (short)0, c, false, false);
}

// A-fragment (16x32 f16) from an fp32 row: lane holds row M=lane%16.
__device__ __forceinline__ v16h load_a_f32(const float* __restrict__ rowptr, int k0, int h) {
    const float4* p0 = (const float4*)(rowptr + k0 + 16 * h);
    const float4* p1 = (const float4*)(rowptr + k0 + 16 * h + 8);
    float4 f0 = p0[0], f1 = p0[1], f2 = p1[0], f3 = p1[1];
    v16h a;
    a[0] = (_Float16)f0.x; a[1] = (_Float16)f0.y; a[2] = (_Float16)f0.z; a[3] = (_Float16)f0.w;
    a[4] = (_Float16)f1.x; a[5] = (_Float16)f1.y; a[6] = (_Float16)f1.z; a[7] = (_Float16)f1.w;
    a[8] = (_Float16)f2.x; a[9] = (_Float16)f2.y; a[10] = (_Float16)f2.z; a[11] = (_Float16)f2.w;
    a[12] = (_Float16)f3.x; a[13] = (_Float16)f3.y; a[14] = (_Float16)f3.z; a[15] = (_Float16)f3.w;
    return a;
}

// 16 contiguous halves (generic pointer: global or LDS), 16B-aligned.
__device__ __forceinline__ v16h load_h16(const _Float16* p) {
    F16x8 a0, a1;
    a0.u = *(const uint4*)(p);
    a1.u = *(const uint4*)(p + 8);
    v16h a;
#pragma unroll
    for (int i = 0; i < 8; ++i) { a[i] = a0.h[i]; a[8 + i] = a1.h[i]; }
    return a;
}

// A-fragment from contiguous f16 row (two 16B segments).
__device__ __forceinline__ v16h load_a_f16(const _Float16* __restrict__ rowptr, int k0, int h) {
    F16x8 a0, a1;
    a0.u = *(const uint4*)(rowptr + k0 + 16 * h);
    a1.u = *(const uint4*)(rowptr + k0 + 16 * h + 8);
    v16h a;
#pragma unroll
    for (int i = 0; i < 8; ++i) { a[i] = a0.h[i]; a[8 + i] = a1.h[i]; }
    return a;
}

// ---------------------------------------------------------------------------
// Kernel 0: convert W to f16 once; 1/sqrt(448) folded into Wq.
// ---------------------------------------------------------------------------
__global__ void __launch_bounds__(256) wconv_kernel(
    const float* __restrict__ Wk, const float* __restrict__ Wq, const float* __restrict__ Wv,
    _Float16* __restrict__ wk16, _Float16* __restrict__ wq16, _Float16* __restrict__ wv16)
{
    const int i = blockIdx.x * 256 + threadIdx.x;
    if (i < H_SZ * C_SZ) {
        wk16[i] = (_Float16)Wk[i];
        wq16[i] = (_Float16)(Wq[i] * 0.047245559f);
        wv16[i] = (_Float16)Wv[i];
    }
}

// ---------------------------------------------------------------------------
// Kernel 1: q/k row-major (B,T,64) f16; v stored transposed (B,64,T) f16.
// ---------------------------------------------------------------------------
__global__ void __launch_bounds__(256) proj_kernel(
    const float* __restrict__ x,
    const _Float16* __restrict__ wq16, const _Float16* __restrict__ wk16,
    const _Float16* __restrict__ wv16,
    _Float16* __restrict__ qf, _Float16* __restrict__ kf, _Float16* __restrict__ vtf)
{
    const int tid = threadIdx.x;
    const int w   = tid >> 5;
    const int lane = tid & 31;
    const int l16 = lane & 15;
    const int h   = lane >> 4;
    const int bb  = blockIdx.x;
    const int b   = bb >> 4;
    const int R0  = (bb & 15) * 128 + w * 16;

    const float* xrow = x + ((size_t)b * T_SZ + R0 + l16) * C_SZ;

    v8f aq[4], ak[4], av[4];
#pragma unroll
    for (int i = 0; i < 4; ++i) { v8f z = {}; aq[i] = z; ak[i] = z; av[i] = z; }

    for (int kc = 0; kc < C_SZ / 32; ++kc) {
        const int k0 = kc * 32;
        v16h ax = load_a_f32(xrow, k0, h);
        v16h bq[4], bk[4], bv[4];
#pragma unroll
        for (int nb = 0; nb < 4; ++nb) {
            const size_t wof = (size_t)(nb * 16 + l16) * C_SZ + k0 + 16 * h;
            bq[nb] = load_h16(wq16 + wof);
            bk[nb] = load_h16(wk16 + wof);
            bv[nb] = load_h16(wv16 + wof);
        }
#pragma unroll
        for (int nb = 0; nb < 4; ++nb) {
            aq[nb] = wmma16(ax, bq[nb], aq[nb]);
            ak[nb] = wmma16(ax, bk[nb], ak[nb]);
            av[nb] = wmma16(ax, bv[nb], av[nb]);
        }
    }

    const size_t rbase = (size_t)b * T_SZ + R0;
#pragma unroll
    for (int nb = 0; nb < 4; ++nb) {
#pragma unroll
        for (int r = 0; r < 8; ++r) {
            const size_t idx = (rbase + r + 8 * h) * H_SZ + nb * 16 + l16;
            qf[idx] = (_Float16)(aq[nb][r]);
            kf[idx] = (_Float16)(ak[nb][r]);
        }
        F16x8 pk;
#pragma unroll
        for (int r = 0; r < 8; ++r) pk.h[r] = (_Float16)(av[nb][r]);
        const size_t vidx = ((size_t)b * H_SZ + nb * 16 + l16) * T_SZ + R0 + 8 * h;
        *(uint4*)(vtf + vidx) = pk.u;
    }
}

// ---------------------------------------------------------------------------
// Kernel 2: causal flash attention with double-buffered async K/V^T tiles:
// async DMA for block jb+1 overlaps WMMA/softmax of block jb.
// ---------------------------------------------------------------------------
__global__ void __launch_bounds__(256) attn_kernel(
    const _Float16* __restrict__ qf, const _Float16* __restrict__ kf,
    const _Float16* __restrict__ vtf, float* __restrict__ out)
{
    __shared__ __align__(16) _Float16 kt[2][64][72];     // [buf][key][head dim]
    __shared__ __align__(16) _Float16 vt[2][64][72];     // [buf][head dim][key]
    __shared__ __align__(16) _Float16 plds[8][16][72];

    const int tid = threadIdx.x;
    const int w   = tid >> 5;
    const int lane = tid & 31;
    const int l16 = lane & 15;
    const int h   = lane >> 4;
    const int bb  = blockIdx.x;
    const int b   = bb >> 4;
    const int rb  = bb & 15;
    const int R0  = rb * 128 + w * 16;

    const _Float16* qrow = qf + ((size_t)b * T_SZ + R0 + l16) * H_SZ;
    v16h qa0 = load_a_f16(qrow, 0, h);
    v16h qa1 = load_a_f16(qrow, 32, h);

    float m8[8], l8[8];
#pragma unroll
    for (int r = 0; r < 8; ++r) { m8[r] = -1.0e30f; l8[r] = 0.0f; }
    v8f o[4];
#pragma unroll
    for (int i = 0; i < 4; ++i) { v8f z = {}; o[i] = z; }

    const int nbmax = 2 * rb + 2;          // covers rows up to R0_base+127
    const int frow = tid >> 2;             // cooperative fill: 64 rows
    const int fseg = tid & 3;              // 4 x 16-half segments per row

    // issue tiles for key block `kb0` into buffer `buf`
    auto issue_tiles = [&](int kb0, int buf) {
        const _Float16* gk = kf  + ((size_t)b * T_SZ + kb0 + frow) * H_SZ + fseg * 16;
        const _Float16* gv = vtf + ((size_t)b * H_SZ + frow) * T_SZ + kb0 + fseg * 16;
        _Float16* lk = &kt[buf][frow][fseg * 16];
        _Float16* lv = &vt[buf][frow][fseg * 16];
        cp16_g2l(gk,     lk);
        cp16_g2l(gk + 8, lk + 8);
        cp16_g2l(gv,     lv);
        cp16_g2l(gv + 8, lv + 8);
    };

    issue_tiles(0, 0);                     // prologue: prefetch block 0

    for (int jb = 0; jb < nbmax; ++jb) {
        const int kb0 = jb * 64;
        const int cur = jb & 1;

        wait_g2l();                        // my copies for buf[cur] done
        __syncthreads();                   // everyone's copies done / readers of other buf done
        if (jb + 1 < nbmax) issue_tiles(kb0 + 64, cur ^ 1);   // overlap with compute

        if (kb0 < R0 + 16) {               // causal cutoff for this wave
            // ---- S = Q * K^T ----
            v8f s[4];
#pragma unroll
            for (int t = 0; t < 4; ++t) {
                const _Float16* krow = &kt[cur][t * 16 + l16][0];
                v16h b0 = load_h16(krow + 16 * h);
                v16h b1 = load_h16(krow + 32 + 16 * h);
                v8f z = {};
                v8f acc = wmma16(qa0, b0, z);
                s[t] = wmma16(qa1, b1, acc);
            }

            // ---- causal mask (diagonal block only) ----
            if (kb0 + 63 > R0) {
#pragma unroll
                for (int t = 0; t < 4; ++t)
#pragma unroll
                    for (int r = 0; r < 8; ++r) {
                        const int col = kb0 + t * 16 + l16;
                        const int row = R0 + r + 8 * h;
                        if (col > row) s[t][r] = -1.0e30f;
                    }
            }

            // ---- online softmax ----
            float alpha[8];
#pragma unroll
            for (int r = 0; r < 8; ++r) {
                float rm = fmaxf(fmaxf(s[0][r], s[1][r]), fmaxf(s[2][r], s[3][r]));
                rm = fmaxf(rm, __shfl_xor(rm, 1, 32));
                rm = fmaxf(rm, __shfl_xor(rm, 2, 32));
                rm = fmaxf(rm, __shfl_xor(rm, 4, 32));
                rm = fmaxf(rm, __shfl_xor(rm, 8, 32));
                const float mn = fmaxf(m8[r], rm);
                const float al = __expf(m8[r] - mn);
                float rs = 0.0f;
#pragma unroll
                for (int t = 0; t < 4; ++t) {
                    const float pv = __expf(s[t][r] - mn);
                    s[t][r] = pv;
                    rs += pv;
                }
                rs += __shfl_xor(rs, 1, 32);
                rs += __shfl_xor(rs, 2, 32);
                rs += __shfl_xor(rs, 4, 32);
                rs += __shfl_xor(rs, 8, 32);
                l8[r] = l8[r] * al + rs;
                m8[r] = mn;
                alpha[r] = al;
            }
#pragma unroll
            for (int nb = 0; nb < 4; ++nb)
#pragma unroll
                for (int r = 0; r < 8; ++r) o[nb][r] *= alpha[r];

            // ---- P: C-layout -> per-wave LDS -> A-layout ----
#pragma unroll
            for (int t = 0; t < 4; ++t)
#pragma unroll
                for (int r = 0; r < 8; ++r)
                    plds[w][r + 8 * h][t * 16 + l16] = (_Float16)s[t][r];

#pragma unroll
            for (int c = 0; c < 2; ++c) {
                const _Float16* prow = &plds[w][l16][0];
                v16h pa = load_a_f16(prow, c * 32, h);
#pragma unroll
                for (int nb = 0; nb < 4; ++nb) {
                    v16h bv = load_h16(&vt[cur][nb * 16 + l16][c * 32 + 16 * h]);
                    o[nb] = wmma16(pa, bv, o[nb]);
                }
            }
        }
    }

    // ---- epilogue: out = O / l ----
#pragma unroll
    for (int r = 0; r < 8; ++r) {
        const float invl = 1.0f / l8[r];
        const size_t row = (size_t)b * T_SZ + R0 + r + 8 * h;
#pragma unroll
        for (int nb = 0; nb < 4; ++nb)
            out[row * H_SZ + nb * 16 + l16] = o[nb][r] * invl;
    }
}

extern "C" void kernel_launch(void* const* d_in, const int* in_sizes, int n_in,
                              void* d_out, int out_size, void* d_ws, size_t ws_size,
                              hipStream_t stream) {
    (void)in_sizes; (void)n_in; (void)out_size; (void)ws_size;
    const float* x  = (const float*)d_in[0];
    const float* Wk = (const float*)d_in[1];
    const float* Wq = (const float*)d_in[2];
    const float* Wv = (const float*)d_in[3];
    float* out = (float*)d_out;

    const size_t elems = (size_t)B_SZ * T_SZ * H_SZ;   // 2M
    const size_t welems = (size_t)H_SZ * C_SZ;         // 28672
    _Float16* qf   = (_Float16*)d_ws;
    _Float16* kff  = qf + elems;
    _Float16* vtf  = kff + elems;
    _Float16* wk16 = vtf + elems;
    _Float16* wq16 = wk16 + welems;
    _Float16* wv16 = wq16 + welems;

    wconv_kernel<<<dim3((int)((welems + 255) / 256)), dim3(256), 0, stream>>>(
        Wk, Wq, Wv, wk16, wq16, wv16);

    const int nblk = (B_SZ * T_SZ) / 128;              // 256 workgroups
    proj_kernel<<<dim3(nblk), dim3(256), 0, stream>>>(x, wq16, wk16, wv16, qf, kff, vtf);
    attn_kernel<<<dim3(nblk), dim3(256), 0, stream>>>(qf, kff, vtf, out);
}
